// MoE_1443109011689
// MI455X (gfx1250) — matrix operand order repaired and checked by reading, not compiled, not run on hardware
//
#include <hip/hip_runtime.h>
#include <hip/hip_bf16.h>

// Problem constants (from reference): B=2, S=2048, D=1024, HD=2048, E=8, K=2
#define NT   4096   // tokens = B*S
#define DD   1024   // model dim
#define HDIM 2048   // hidden dim
#define NE   8      // experts
#define KSEL 2      // top-k
#define LS   48     // LDS row stride in bf16 elements (96B, 16B-aligned rows)

typedef __attribute__((ext_vector_type(16))) __bf16 v16bf;
typedef __attribute__((ext_vector_type(8)))  float  v8f;
typedef __attribute__((ext_vector_type(4)))  unsigned v4u;
typedef __attribute__((ext_vector_type(8)))  unsigned v8u;

union Frag  { v16bf v; uint4 q[2]; };
union Pack8 { uint4 q; unsigned short h[8]; unsigned w[4]; };

__device__ __forceinline__ unsigned short f2bf(float f) {
    union { float f; unsigned u; } c; c.f = f;
    unsigned r = c.u + 0x7FFFu + ((c.u >> 16) & 1u);   // RNE
    return (unsigned short)(r >> 16);
}

// Packed f32x2 -> bf16x2: hardware packed cvt (probe-verified to assemble on
// gfx1250; clang builtin not declared on this toolchain).
__device__ __forceinline__ unsigned pk2bf(float a, float b) {
#if __has_builtin(__builtin_amdgcn_cvt_pk_bf16_f32)
    auto r = __builtin_amdgcn_cvt_pk_bf16_f32(a, b);
    unsigned u;
    __builtin_memcpy(&u, &r, 4);
    return u;
#else
    unsigned u;
    asm("v_cvt_pk_bf16_f32 %0, %1, %2" : "=v"(u) : "v"(a), "v"(b));
    return u;
#endif
}

__device__ __forceinline__ uint4 pack8f(float4 u0, float4 u1) {
    Pack8 pk;
    pk.w[0] = pk2bf(u0.x, u0.y);
    pk.w[1] = pk2bf(u0.z, u0.w);
    pk.w[2] = pk2bf(u1.x, u1.y);
    pk.w[3] = pk2bf(u1.z, u1.w);
    return pk.q;
}

// Async global->LDS 16B copy (CDNA5): per-lane LDS dest addr + global src addr.
// Tracked with ASYNCcnt; caller must s_wait_asynccnt before the barrier.
__device__ __forceinline__ void async_copy_b128(void* lds_dst, const void* gsrc) {
    unsigned ldsa = (unsigned)(size_t)lds_dst;            // low 32b = LDS byte offset
    unsigned long long ga = (unsigned long long)(size_t)gsrc;
    asm volatile("global_load_async_to_lds_b128 %0, %1, off"
                 :: "v"(ldsa), "v"(ga) : "memory");
}
__device__ __forceinline__ void wait_async0() {
    asm volatile("s_wait_asynccnt 0x0" ::: "memory");
}

// Tensor Data Mover: one 2D tile load (global -> LDS), D# groups 0/1 in SGPRs,
// groups 2/3 NULL (<=2D tensor). Tracked with TENSORcnt.
// g1w0 packs: data_size | pad_enable | pad_interval | pad_amount.
__device__ __forceinline__ void tdm_load_2d(unsigned lds_addr, const void* gsrc,
                                            unsigned g1w0,
                                            unsigned tdim0, unsigned tdim1,
                                            unsigned tile0, unsigned tile1,
                                            unsigned stride0) {
    unsigned long long ga = (unsigned long long)(size_t)gsrc;
    v4u g0;
    g0[0] = 1u;                                            // count=1 valid descriptor
    g0[1] = lds_addr;                                      // lds_addr [63:32]
    g0[2] = (unsigned)ga;                                  // global_addr [95:64]
    g0[3] = (unsigned)((ga >> 32) & 0x01FFFFFFu)           // global_addr [120:96]
          | (2u << 30);                                    // type=2 ("image")
    v8u g1;
    g1[0] = g1w0;                                          // mask=0, data_size, pad cfg
    g1[1] = (tdim0 & 0xFFFFu) << 16;                       // tensor_dim0 [63:48]
    g1[2] = ((tdim0 >> 16) & 0xFFFFu)                      // tensor_dim0 [79:64]
          | ((tdim1 & 0xFFFFu) << 16);                     // tensor_dim1 [95:80]
    g1[3] = ((tdim1 >> 16) & 0xFFFFu)                      // tensor_dim1 [111:96]
          | ((tile0 & 0xFFFFu) << 16);                     // tile_dim0  [127:112]
    g1[4] = tile1 & 0xFFFFu;                               // tile_dim1 (tile_dim2=0)
    g1[5] = stride0;                                       // tensor_dim0_stride [191:160]
    g1[6] = 0;                                             // stride hi + dim1_stride lo
    g1[7] = 0;
    asm volatile("tensor_load_to_lds %0, %1" :: "s"(g0), "s"(g1) : "memory");
}

// ---------------- Kernel 1: zero output + workspace header ----------------
__global__ void moe_zero(float* __restrict__ out, int n_out, int* __restrict__ hdr) {
    long i = (long)blockIdx.x * blockDim.x + threadIdx.x;
    if (i < n_out) out[i] = 0.0f;
    if (blockIdx.x == 0 && threadIdx.x < 64) hdr[threadIdx.x] = 0;  // counts/psum/base
}

// ---------------- Kernel 2: router (one wave32 per token) ----------------
__global__ __launch_bounds__(256) void moe_router(
        const float* __restrict__ x, const float* __restrict__ Wg,
        int* __restrict__ counts, float* __restrict__ psum,
        int* __restrict__ tok_list, float* __restrict__ gate_list) {
    __shared__ float wg_s[NE * DD];           // 32 KB
    int tid = threadIdx.x;
    for (int i = tid; i < NE * DD; i += 256) wg_s[i] = Wg[i];
    __syncthreads();

    int wid = tid >> 5, lane = tid & 31;
    int n = blockIdx.x * 8 + wid;
    const float* xr = x + (size_t)n * DD;

    float part[NE];
#pragma unroll
    for (int e = 0; e < NE; e++) part[e] = 0.0f;
#pragma unroll
    for (int i = 0; i < 8; i++) {
        int d = i * 128 + lane * 4;
        float4 xv = *(const float4*)(xr + d);
#pragma unroll
        for (int e = 0; e < NE; e++) {
            float4 wv = *(const float4*)(&wg_s[e * DD + d]);
            part[e] += xv.x * wv.x + xv.y * wv.y + xv.z * wv.z + xv.w * wv.w;
        }
    }
#pragma unroll
    for (int e = 0; e < NE; e++) {
#pragma unroll
        for (int off = 16; off > 0; off >>= 1)
            part[e] += __shfl_xor(part[e], off, 32);
    }
    if (lane == 0) {
        float mx = part[0];
        for (int e = 1; e < NE; e++) mx = fmaxf(mx, part[e]);
        float p[NE]; float s = 0.0f;
        for (int e = 0; e < NE; e++) { p[e] = __expf(part[e] - mx); s += p[e]; }
        float inv = 1.0f / s;
        for (int e = 0; e < NE; e++) { p[e] *= inv; atomicAdd(&psum[e], p[e]); }
        int e1 = 0; float v1 = p[0];
        for (int e = 1; e < NE; e++) if (p[e] > v1) { v1 = p[e]; e1 = e; }
        int e2 = -1; float v2 = -1.0f;
        for (int e = 0; e < NE; e++) if (e != e1 && p[e] > v2) { v2 = p[e]; e2 = e; }
        float inv2 = 1.0f / (v1 + v2);
        int p1 = atomicAdd(&counts[e1], 1);
        tok_list[e1 * NT + p1] = n; gate_list[e1 * NT + p1] = v1 * inv2;
        int p2 = atomicAdd(&counts[e2], 1);
        tok_list[e2 * NT + p2] = n; gate_list[e2 * NT + p2] = v2 * inv2;
    }
}

// ---------------- Kernel 3: padded prefix offsets + balancing loss ----------------
__global__ void moe_finalize(const int* __restrict__ counts, const float* __restrict__ psum,
                             int* __restrict__ base, float* __restrict__ out_loss) {
    if (threadIdx.x == 0 && blockIdx.x == 0) {
        int acc = 0; float L = 0.0f;
        for (int e = 0; e < NE; e++) {
            base[e] = acc;
            acc += ((counts[e] + 63) >> 6) << 6;              // pad to 64-row tiles
            L += (psum[e] / (float)NT) * ((float)counts[e] / (float)(NT * KSEL));
        }
        *out_loss = L;
    }
}

// ---------------- Kernel 4: act = silu(x@W1^T) * (x@Wgate^T)  (bf16 WMMA) ----------------
// Block tile: 64 tokens x 128 hidden. 8 waves in 4(M) x 2(N); wave: 16x64 = 4 WMMA tiles x2 streams.
__global__ __launch_bounds__(256) void moe_act(
        const float* __restrict__ x,
        const float* __restrict__ W1, const float* __restrict__ Wgate,
        const int* __restrict__ counts, const int* __restrict__ base,
        const int* __restrict__ tok_list, unsigned short* __restrict__ act) {
    int e = blockIdx.z;
    int count = counts[e];
    int m0 = blockIdx.y * 64;
    if (m0 >= count) return;
    int n0 = blockIdx.x * 128;

    __shared__ int tok_s[64];
    __shared__ unsigned short xs [64  * LS];
    __shared__ unsigned short b1s[128 * LS];
    __shared__ unsigned short bgs[128 * LS];

    int tid = threadIdx.x;
    if (tid < 64) {
        int pos = m0 + tid;
        tok_s[tid] = (pos < count) ? tok_list[e * NT + pos]
                                   : tok_list[e * NT + (count > 0 ? count - 1 : 0)];
    }
    __syncthreads();

    const float* w1p = W1    + (size_t)e * HDIM * DD;
    const float* wgp = Wgate + (size_t)e * HDIM * DD;

    int wid = tid >> 5, lane = tid & 31;
    int wm = wid >> 1, wn = wid & 1;
    v8f acc1[4] = {};
    v8f accg[4] = {};

    int arow = tid >> 2, acseg = (tid & 3) * 8;   // A loader: 64 rows x 32 K, 8 f32 each
    int brow = tid >> 1, bcseg = (tid & 1) * 16;  // B loader: 128 rows x 32 K, 16 f32 each
    int atok = tok_s[arow];

    for (int k0 = 0; k0 < DD; k0 += 32) {
        __syncthreads();
        {   // stage A (gathered x rows, f32 -> bf16)
            const float* src = x + (size_t)atok * DD + k0 + acseg;
            float4 u0 = *(const float4*)(src);
            float4 u1 = *(const float4*)(src + 4);
            *(uint4*)&xs[arow * LS + acseg] = pack8f(u0, u1);
        }
        {   // stage B1 / Bgate (W1, Wgate rows, f32 -> bf16)
            const float* s1 = w1p + (size_t)(n0 + brow) * DD + k0 + bcseg;
            const float* sg = wgp + (size_t)(n0 + brow) * DD + k0 + bcseg;
            if (k0 + 32 < DD) { __builtin_prefetch(s1 + 32); __builtin_prefetch(sg + 32); }
#pragma unroll
            for (int h = 0; h < 2; h++) {
                float4 u0 = *(const float4*)(s1 + h * 8);
                float4 u1 = *(const float4*)(s1 + h * 8 + 4);
                *(uint4*)&b1s[brow * LS + bcseg + h * 8] = pack8f(u0, u1);
                float4 g0 = *(const float4*)(sg + h * 8);
                float4 g1 = *(const float4*)(sg + h * 8 + 4);
                *(uint4*)&bgs[brow * LS + bcseg + h * 8] = pack8f(g0, g1);
            }
        }
        __syncthreads();

        // A fragment 16x32: lane row = l&15; contiguous K = 16*(l>>4) + s
        Frag a;
        int ar = wm * 16 + (lane & 15);
        int kb = (lane >> 4) * 16;
        a.q[0] = *(const uint4*)&xs[ar * LS + kb];
        a.q[1] = *(const uint4*)&xs[ar * LS + kb + 8];
#pragma unroll
        for (int j = 0; j < 4; j++) {
            int col = wn * 64 + j * 16 + (lane & 15);
            int kb2 = (lane >> 4) * 16;
            Frag b1, bg;
            b1.q[0] = *(const uint4*)&b1s[col * LS + kb2];
            b1.q[1] = *(const uint4*)&b1s[col * LS + kb2 + 8];
            bg.q[0] = *(const uint4*)&bgs[col * LS + kb2];
            bg.q[1] = *(const uint4*)&bgs[col * LS + kb2 + 8];
            acc1[j] = __builtin_amdgcn_wmma_f32_16x16x32_bf16(
                          false, a.v, false, b1.v, (short)0, acc1[j], false, false);
            accg[j] = __builtin_amdgcn_wmma_f32_16x16x32_bf16(
                          false, a.v, false, bg.v, (short)0, accg[j], false, false);
        }
    }

    // epilogue: silu(h1)*hg, store bf16 act rows (C layout: M = r + 8*(lane>=16), N = lane&15)
    int rowbase = base[e] + m0 + wm * 16 + (lane >> 4) * 8;
    int colbase = n0 + wn * 64 + (lane & 15);
#pragma unroll
    for (int j = 0; j < 4; j++) {
#pragma unroll
        for (int r = 0; r < 8; r++) {
            float h1 = acc1[j][r];
            float hg = accg[j][r];
            float av = h1 / (1.0f + __expf(-h1)) * hg;
            act[(size_t)(rowbase + r) * HDIM + colbase + j * 16] = f2bf(av);
        }
    }
}

// ---------------- Kernel 5: out += gate * (act @ W2^T), atomic scatter ----------------
__global__ __launch_bounds__(256) void moe_out(
        const unsigned short* __restrict__ act, const float* __restrict__ W2,
        const int* __restrict__ counts, const int* __restrict__ base,
        const int* __restrict__ tok_list, const float* __restrict__ gate_list,
        float* __restrict__ out) {
    int e = blockIdx.z;
    int count = counts[e];
    int m0 = blockIdx.y * 64;
    if (m0 >= count) return;
    int n0 = blockIdx.x * 128;

    __shared__ int   tok_s[64];
    __shared__ float gate_s[64];
    __shared__ unsigned short as_[64  * LS];
    __shared__ unsigned short bs_[128 * LS];

    int tid = threadIdx.x;
    if (tid < 64) {
        int pos = m0 + tid;
        bool valid = pos < count;
        tok_s[tid]  = valid ? tok_list [e * NT + pos] : -1;
        gate_s[tid] = valid ? gate_list[e * NT + pos] : 0.0f;
    }
    __syncthreads();

    const float* w2p = W2 + (size_t)e * DD * HDIM;
    const unsigned short* ap = act + (size_t)(base[e] + m0) * HDIM;

    int wid = tid >> 5, lane = tid & 31;
    int wm = wid >> 1, wn = wid & 1;
    v8f acc[4] = {};

    int arow = tid >> 2, acseg = (tid & 3) * 8;
    int brow = tid >> 1, bcseg = (tid & 1) * 16;
    bool tdm_wave = (wid == 0);
    // TDM group-1 word0: workgroup_mask=0, data_size=1 (2B), pad_enable=1,
    // pad_interval=3 (pad after 16 DWORDs), pad_amount=7 (8 DWORDs) -> 96B LDS pitch
    const unsigned g1w0 = (1u << 16) | (1u << 20) | (3u << 22) | (7u << 25);

    for (int k0 = 0; k0 < HDIM; k0 += 32) {
        __syncthreads();
        // stage A rows 0..31: one TDM 2D-tile descriptor (32x32 bf16, stride HDIM)
        if (tdm_wave)
            tdm_load_2d((unsigned)(size_t)&as_[0], ap + k0, g1w0,
                        /*tdim0=*/HDIM, /*tdim1=*/32,
                        /*tile0=*/32, /*tile1=*/32, /*stride0=*/HDIM);
        // stage A rows 32..63: per-lane async 16B copies (threads 128..255)
        if (tid >= 128)
            async_copy_b128(&as_[arow * LS + acseg],
                            ap + (size_t)arow * HDIM + k0 + acseg);
        {   // stage B: W2 rows, f32 -> bf16 (all threads)
            const float* s2 = w2p + (size_t)(n0 + brow) * HDIM + k0 + bcseg;
            if (k0 + 32 < HDIM) __builtin_prefetch(s2 + 32);
#pragma unroll
            for (int h = 0; h < 2; h++) {
                float4 u0 = *(const float4*)(s2 + h * 8);
                float4 u1 = *(const float4*)(s2 + h * 8 + 4);
                *(uint4*)&bs_[brow * LS + bcseg + h * 8] = pack8f(u0, u1);
            }
        }
        wait_async0();                                   // drain ASYNCcnt (per-wave)
        if (tdm_wave) __builtin_amdgcn_s_wait_tensorcnt(0);  // drain TENSORcnt
        __syncthreads();

        Frag a;
        int ar = wm * 16 + (lane & 15);
        int kb = (lane >> 4) * 16;
        a.q[0] = *(const uint4*)&as_[ar * LS + kb];
        a.q[1] = *(const uint4*)&as_[ar * LS + kb + 8];
#pragma unroll
        for (int j = 0; j < 4; j++) {
            int col = wn * 64 + j * 16 + (lane & 15);
            int kb2 = (lane >> 4) * 16;
            Frag b;
            b.q[0] = *(const uint4*)&bs_[col * LS + kb2];
            b.q[1] = *(const uint4*)&bs_[col * LS + kb2 + 8];
            acc[j] = __builtin_amdgcn_wmma_f32_16x16x32_bf16(
                         false, a.v, false, b.v, (short)0, acc[j], false, false);
        }
    }

    int mloc = wm * 16 + (lane >> 4) * 8;
    int colbase = n0 + wn * 64 + (lane & 15);
#pragma unroll
    for (int j = 0; j < 4; j++) {
#pragma unroll
        for (int r = 0; r < 8; r++) {
            int m = mloc + r;
            int tok = tok_s[m];
            if (tok >= 0)
                atomicAdd(&out[(size_t)tok * DD + colbase + j * 16],
                          gate_s[m] * acc[j][r]);
        }
    }
}

// ---------------- host launch ----------------
extern "C" void kernel_launch(void* const* d_in, const int* in_sizes, int n_in,
                              void* d_out, int out_size, void* d_ws, size_t ws_size,
                              hipStream_t stream) {
    (void)in_sizes; (void)n_in; (void)out_size; (void)ws_size;
    const float* x     = (const float*)d_in[0];
    const float* Wg    = (const float*)d_in[1];
    const float* W1    = (const float*)d_in[2];
    const float* Wgate = (const float*)d_in[3];
    const float* W2    = (const float*)d_in[4];
    float* out = (float*)d_out;

    char* ws = (char*)d_ws;
    int*   counts    = (int*)ws;                           // 8 ints @ 0
    float* psum      = (float*)(ws + 64);                  // 8 floats @ 64
    int*   base      = (int*)(ws + 128);                   // 8 ints @ 128
    int*   tok_list  = (int*)(ws + 256);                   // E*NT ints
    float* gate_list = (float*)(ws + 256 + NE * NT * 4);   // E*NT floats
    unsigned short* act = (unsigned short*)(ws + (1 << 19)); // bf16 act rows

    int n_out = NT * DD + 1;
    moe_zero<<<(n_out + 255) / 256, 256, 0, stream>>>(out, n_out, (int*)ws);
    moe_router<<<NT / 8, 256, 0, stream>>>(x, Wg, counts, psum, tok_list, gate_list);
    moe_finalize<<<1, 64, 0, stream>>>(counts, psum, base, out + (size_t)NT * DD);
    moe_act<<<dim3(HDIM / 128, NT / 64, NE), 256, 0, stream>>>(
        x, W1, Wgate, counts, base, tok_list, act);
    moe_out<<<dim3(DD / 128, NT / 64, NE), 256, 0, stream>>>(
        act, W2, counts, base, tok_list, gate_list, out);
}
